// MambaBlock_7361573945747
// MI455X (gfx1250) — matrix operand order repaired and checked by reading, not compiled, not run on hardware
//
#include <hip/hip_runtime.h>

#define D_MODEL 1024
#define D_STATE 16
#define D_CONV  4
#define D_INNER 2048
#define DT_RANK 64
#define B_SZ    2
#define T_LEN   1024
#define MROWS   (B_SZ * T_LEN)          // 2048 token rows
#define N_XZ    (2 * D_INNER)           // 4096
#define N_DBL   (DT_RANK + 2 * D_STATE) // 96

typedef __bf16 bf16_t;
typedef bf16_t v16bf __attribute__((ext_vector_type(16)));
typedef bf16_t v8bf  __attribute__((ext_vector_type(8)));
typedef float  v8f   __attribute__((ext_vector_type(8)));

union AB16 { v8bf h2[2]; v16bf full; };
union B16U { v8bf q[2]; v16bf full; };

#define AS1 __attribute__((address_space(1)))
#define AS3 __attribute__((address_space(3)))

// 16-byte payload type matching the async-LDS builtin's expected pointee
typedef int b128_t __attribute__((vector_size(16)));

#if defined(__has_builtin)
#if __has_builtin(__builtin_amdgcn_global_load_async_to_lds_b128) && \
    __has_builtin(__builtin_amdgcn_s_wait_asynccnt)
#define USE_ASYNC_LDS 1
#endif
#endif

#ifdef USE_ASYNC_LDS
// async 16B copy: global -> LDS, tracked by ASYNCcnt
__device__ __forceinline__ void async_cp16(const void* g, void* l) {
    __builtin_amdgcn_global_load_async_to_lds_b128(
        (AS1 b128_t*)(AS1 void*)(void*)g,
        (AS3 b128_t*)(AS3 void*)(void*)l, 0, 0);
}
#endif

// ---------------------------------------------------------------- f32 -> bf16
__global__ __launch_bounds__(256)
void cvt_f32_to_bf16(const float* __restrict__ src, bf16_t* __restrict__ dst, int n) {
    int i = blockIdx.x * blockDim.x + threadIdx.x;
    if (i < n) dst[i] = (bf16_t)src[i];
}

// ===================== LDS-tiled bf16 WMMA GEMM: C = A * W^T =================
// Block: 256 threads (8 waves) -> 128x128 tile. Wave grid 4(M) x 2(N); each
// wave owns 32x64 (2x4 WMMA subtiles). K staged in 32-deep chunks, double-
// buffered in LDS via GLOBAL_LOAD_ASYNC_TO_LDS_B128 (ASYNCcnt) when available.
#define BM  128
#define BN  128
#define LDP 40   // LDS row pitch in bf16 (80B): 16B-aligned, conflict-free

__global__ __launch_bounds__(256)
void gemm_lds_wmma(const bf16_t* __restrict__ A,   // [M,K] row-major
                   const bf16_t* __restrict__ W,   // [N,K] row-major
                   float* __restrict__ C,          // [M,ldc]
                   int K, int ldc) {
    __shared__ bf16_t sA[2][BM * LDP];
    __shared__ bf16_t sB[2][BN * LDP];

    const int tid  = threadIdx.x;
    const int lane = tid & 31;
    const int w    = tid >> 5;
    const int wm   = w & 3;    // wave row 0..3
    const int wn   = w >> 2;   // wave col 0..1
    const int r    = lane & 15;
    const int h    = lane >> 4;

    const int m0 = blockIdx.y * BM;
    const int n0 = blockIdx.x * BN;

    // staging: 512 16B segments per tile; each thread owns segments tid*2, tid*2+1
    const int s0  = tid * 2, s1 = tid * 2 + 1;
    const int r0_ = s0 >> 2, k0_ = (s0 & 3) * 8;
    const int r1_ = s1 >> 2, k1_ = (s1 & 3) * 8;

    auto stage = [&](int buf, int kk) {
#ifdef USE_ASYNC_LDS
        async_cp16(A + (size_t)(m0 + r0_) * K + kk + k0_, &sA[buf][r0_ * LDP + k0_]);
        async_cp16(A + (size_t)(m0 + r1_) * K + kk + k1_, &sA[buf][r1_ * LDP + k1_]);
        async_cp16(W + (size_t)(n0 + r0_) * K + kk + k0_, &sB[buf][r0_ * LDP + k0_]);
        async_cp16(W + (size_t)(n0 + r1_) * K + kk + k1_, &sB[buf][r1_ * LDP + k1_]);
#else
        *(v8bf*)&sA[buf][r0_ * LDP + k0_] = *(const v8bf*)(A + (size_t)(m0 + r0_) * K + kk + k0_);
        *(v8bf*)&sA[buf][r1_ * LDP + k1_] = *(const v8bf*)(A + (size_t)(m0 + r1_) * K + kk + k1_);
        *(v8bf*)&sB[buf][r0_ * LDP + k0_] = *(const v8bf*)(W + (size_t)(n0 + r0_) * K + kk + k0_);
        *(v8bf*)&sB[buf][r1_ * LDP + k1_] = *(const v8bf*)(W + (size_t)(n0 + r1_) * K + kk + k1_);
#endif
    };

    const v8f vzero = {0.f, 0.f, 0.f, 0.f, 0.f, 0.f, 0.f, 0.f};
    v8f acc[2][4];
#pragma unroll
    for (int i = 0; i < 2; ++i)
#pragma unroll
        for (int j = 0; j < 4; ++j) acc[i][j] = vzero;

    auto compute = [&](int buf) {
#pragma unroll
        for (int i = 0; i < 2; ++i) {
            // A 16x32 fragment: lane l<16 row l K={0..15};
            // lane l+16 row l K={16..31}.
            const bf16_t* ap = &sA[buf][(wm * 32 + i * 16 + r) * LDP];
            AB16 a;
            a.h2[0] = *(const v8bf*)(ap + h * 16);
            a.h2[1] = *(const v8bf*)(ap + h * 16 + 8);
#pragma unroll
            for (int j = 0; j < 4; ++j) {
                // B 32x16 fragment: lane holds col n=r, K = h*16 .. h*16+15
                const bf16_t* bp = &sB[buf][(wn * 64 + j * 16 + r) * LDP + h * 16];
                B16U b;
                b.q[0] = *(const v8bf*)bp;
                b.q[1] = *(const v8bf*)(bp + 8);
                acc[i][j] = __builtin_amdgcn_wmma_f32_16x16x32_bf16(
                    false, a.full, false, b.full, (short)0, acc[i][j], false, false);
            }
        }
    };

    const int nchunks = K >> 5;
#ifdef USE_ASYNC_LDS
    stage(0, 0);
    for (int c = 0; c < nchunks; ++c) {
        __builtin_amdgcn_s_wait_asynccnt(0);
        __syncthreads();                       // chunk c fully staged, prev compute done
        if (c + 1 < nchunks) stage((c + 1) & 1, (c + 1) * 32);
        compute(c & 1);
    }
#else
    for (int c = 0; c < nchunks; ++c) {
        __syncthreads();
        stage(0, c * 32);
        __syncthreads();
        compute(0);
    }
#endif

    // C layout: VGPR ii -> row h*8+ii, col r
#pragma unroll
    for (int i = 0; i < 2; ++i)
#pragma unroll
        for (int j = 0; j < 4; ++j) {
            const int n    = n0 + wn * 64 + j * 16 + r;
            const int mrow = m0 + wm * 32 + i * 16 + h * 8;
#pragma unroll
            for (int ii = 0; ii < 8; ++ii)
                C[(size_t)(mrow + ii) * ldc + n] = acc[i][j][ii];
        }
}

// ================= direct (small) bf16 WMMA GEMM: C = A * W^T ================
template <int NT>
__global__ __launch_bounds__(256)
void gemm_bf16_wmma(const bf16_t* __restrict__ A,
                    const bf16_t* __restrict__ W,
                    float* __restrict__ C,
                    int M, int N, int K, int ldc) {
    const int lane   = threadIdx.x & 31;
    const int wave   = blockIdx.x * (blockDim.x >> 5) + (threadIdx.x >> 5);
    const int tilesN = N / (16 * NT);
    const int tilesM = M / 16;
    if (wave >= tilesM * tilesN) return;
    const int tm = wave / tilesN;
    const int tn = wave % tilesN;
    const int r  = lane & 15;
    const int h  = lane >> 4;

    const bf16_t* Arow = A + (size_t)(tm * 16 + r) * K;

    const v8f vzero = {0.f, 0.f, 0.f, 0.f, 0.f, 0.f, 0.f, 0.f};
    v8f acc[NT];
#pragma unroll
    for (int j = 0; j < NT; ++j) acc[j] = vzero;

    for (int kk = 0; kk < K; kk += 32) {
        AB16 a;
        a.h2[0] = *(const v8bf*)(Arow + kk + h * 16);
        a.h2[1] = *(const v8bf*)(Arow + kk + h * 16 + 8);
#pragma unroll
        for (int j = 0; j < NT; ++j) {
            const bf16_t* Wp = W + (size_t)((tn * NT + j) * 16 + r) * K + kk + h * 16;
            B16U b;
            b.q[0] = *(const v8bf*)Wp;
            b.q[1] = *(const v8bf*)(Wp + 8);
            acc[j] = __builtin_amdgcn_wmma_f32_16x16x32_bf16(
                false, a.full, false, b.full, (short)0, acc[j], false, false);
        }
    }
#pragma unroll
    for (int j = 0; j < NT; ++j) {
        const int n = (tn * NT + j) * 16 + r;
#pragma unroll
        for (int i = 0; i < 8; ++i)
            C[(size_t)(tm * 16 + h * 8 + i) * ldc + n] = acc[j][i];
    }
}

// ----------------------------------- depthwise causal conv (k=4) + bias + SiLU
__global__ __launch_bounds__(256)
void conv1d_silu(const float* __restrict__ xz,
                 const float* __restrict__ conv_w,
                 const float* __restrict__ conv_b,
                 float* __restrict__ xconv,
                 bf16_t* __restrict__ xconv_b) {
    int idx = blockIdx.x * blockDim.x + threadIdx.x;
    if (idx >= MROWS * D_INNER) return;
    int m = idx >> 11;
    int d = idx & (D_INNER - 1);
    int t = m & (T_LEN - 1);
    float acc = conv_b[d];
#pragma unroll
    for (int k = 0; k < D_CONV; ++k) {
        int tt = t - (D_CONV - 1) + k;
        if (tt >= 0)
            acc += conv_w[d * D_CONV + k] * xz[(size_t)(m - (D_CONV - 1) + k) * N_XZ + d];
    }
    float s = acc / (1.f + __expf(-acc));
    xconv[idx]   = s;
    xconv_b[idx] = (bf16_t)s;
}

// ----------------------------------- split x_dbl into dt_low(bf16), B, C (f32)
__global__ __launch_bounds__(256)
void split_xdbl(const float* __restrict__ xdbl,
                bf16_t* __restrict__ dtlow_b,
                float* __restrict__ Bssm,
                float* __restrict__ Cssm) {
    int idx = blockIdx.x * blockDim.x + threadIdx.x;
    if (idx >= MROWS * N_DBL) return;
    int m = idx / N_DBL;
    int j = idx % N_DBL;
    float v = xdbl[idx];
    if (j < DT_RANK)                 dtlow_b[m * DT_RANK + j] = (bf16_t)v;
    else if (j < DT_RANK + D_STATE)  Bssm[m * D_STATE + (j - DT_RANK)] = v;
    else                             Cssm[m * D_STATE + (j - DT_RANK - D_STATE)] = v;
}

// --------------------------------------------------- dt = softplus(raw + b_dt)
__global__ __launch_bounds__(256)
void softplus_dt(float* __restrict__ dt, const float* __restrict__ b_dt) {
    int idx = blockIdx.x * blockDim.x + threadIdx.x;
    if (idx >= MROWS * D_INNER) return;
    int d = idx & (D_INNER - 1);
    float v = dt[idx] + b_dt[d];
    dt[idx] = (v > 20.f) ? v : log1pf(__expf(v));
}

// --------------------------------------------------------- selective SSM scan
#define SCAN_CH 64
__global__ __launch_bounds__(256)
void ssm_scan(const float* __restrict__ dt,
              const float* __restrict__ Bssm,
              const float* __restrict__ Cssm,
              const float* __restrict__ xconv,
              const float* __restrict__ xz,       // z = xz[:, D_INNER:]
              const float* __restrict__ A_log,
              const float* __restrict__ D_param,
              bf16_t* __restrict__ y_b) {
    __shared__ float sB[SCAN_CH][D_STATE];
    __shared__ float sC[SCAN_CH][D_STATE];
    const int b = blockIdx.x >> 3;
    const int d = ((blockIdx.x & 7) << 8) + threadIdx.x;

    float Arow[D_STATE], hst[D_STATE];
#pragma unroll
    for (int n = 0; n < D_STATE; ++n) {
        Arow[n] = -__expf(A_log[d * D_STATE + n]);
        hst[n]  = 0.f;
    }
    const float  Dp   = D_param[d];
    const size_t row0 = (size_t)b * T_LEN;

    for (int t0 = 0; t0 < T_LEN; t0 += SCAN_CH) {
        __syncthreads();
        for (int i = threadIdx.x; i < SCAN_CH * D_STATE; i += blockDim.x) {
            int tt = i >> 4, n = i & 15;
            sB[tt][n] = Bssm[(row0 + t0 + tt) * D_STATE + n];
            sC[tt][n] = Cssm[(row0 + t0 + tt) * D_STATE + n];
        }
        __syncthreads();
        for (int tt = 0; tt < SCAN_CH; ++tt) {
            const size_t m = row0 + t0 + tt;
            __builtin_prefetch(dt    + (m + 4) * D_INNER + d, 0, 1);
            __builtin_prefetch(xconv + (m + 4) * D_INNER + d, 0, 1);
            const float dtv = dt[m * D_INNER + d];
            const float xv  = xconv[m * D_INNER + d];
            float y = 0.f;
#pragma unroll
            for (int n = 0; n < D_STATE; ++n) {
                float dA = __expf(dtv * Arow[n]);
                hst[n] = dA * hst[n] + (dtv * sB[tt][n]) * xv;
                y += hst[n] * sC[tt][n];
            }
            const float zv  = xz[m * N_XZ + D_INNER + d];
            const float out = y * (zv / (1.f + __expf(-zv))) + Dp * xv;
            y_b[m * D_INNER + d] = (bf16_t)out;
        }
    }
}

// ---------------------------------------------------------------------- driver
extern "C" void kernel_launch(void* const* d_in, const int* in_sizes, int n_in,
                              void* d_out, int out_size, void* d_ws, size_t ws_size,
                              hipStream_t stream) {
    const float* x       = (const float*)d_in[0];
    const float* W_in    = (const float*)d_in[1];
    const float* conv_w  = (const float*)d_in[2];
    const float* conv_b  = (const float*)d_in[3];
    const float* W_x     = (const float*)d_in[4];
    const float* W_dt    = (const float*)d_in[5];
    const float* b_dt    = (const float*)d_in[6];
    const float* A_log   = (const float*)d_in[7];
    const float* D_param = (const float*)d_in[8];
    const float* W_out   = (const float*)d_in[9];
    float* out = (float*)d_out;
    (void)in_sizes; (void)n_in; (void)out_size; (void)ws_size;

    char*  ws  = (char*)d_ws;
    size_t off = 0;
    auto alloc = [&](size_t bytes) -> void* {
        void* p = ws + off;
        off += (bytes + 255) & ~(size_t)255;
        return p;
    };

    bf16_t* xb      = (bf16_t*)alloc((size_t)MROWS * D_MODEL * 2);
    bf16_t* Win_b   = (bf16_t*)alloc((size_t)N_XZ * D_MODEL * 2);
    bf16_t* Wx_b    = (bf16_t*)alloc((size_t)N_DBL * D_INNER * 2);
    bf16_t* Wdt_b   = (bf16_t*)alloc((size_t)D_INNER * DT_RANK * 2);
    bf16_t* Wout_b  = (bf16_t*)alloc((size_t)D_MODEL * D_INNER * 2);
    float*  xzbuf   = (float*)alloc((size_t)MROWS * N_XZ * 4);
    float*  xconv   = (float*)alloc((size_t)MROWS * D_INNER * 4);
    bf16_t* xconv_b = (bf16_t*)alloc((size_t)MROWS * D_INNER * 2);
    float*  xdbl    = (float*)alloc((size_t)MROWS * N_DBL * 4);
    bf16_t* dtlow_b = (bf16_t*)alloc((size_t)MROWS * DT_RANK * 2);
    float*  Bssm    = (float*)alloc((size_t)MROWS * D_STATE * 4);
    float*  Cssm    = (float*)alloc((size_t)MROWS * D_STATE * 4);
    float*  dtbuf   = (float*)alloc((size_t)MROWS * D_INNER * 4);   // softplus in place
    bf16_t* y_b     = (bf16_t*)alloc((size_t)MROWS * D_INNER * 2);

    const int TB = 256;
    auto cdiv = [](int a, int b) { return (a + b - 1) / b; };

    // bf16 operand staging
    cvt_f32_to_bf16<<<cdiv(MROWS * D_MODEL, TB), TB, 0, stream>>>(x, xb, MROWS * D_MODEL);
    cvt_f32_to_bf16<<<cdiv(N_XZ * D_MODEL, TB), TB, 0, stream>>>(W_in, Win_b, N_XZ * D_MODEL);
    cvt_f32_to_bf16<<<cdiv(N_DBL * D_INNER, TB), TB, 0, stream>>>(W_x, Wx_b, N_DBL * D_INNER);
    cvt_f32_to_bf16<<<cdiv(D_INNER * DT_RANK, TB), TB, 0, stream>>>(W_dt, Wdt_b, D_INNER * DT_RANK);
    cvt_f32_to_bf16<<<cdiv(D_MODEL * D_INNER, TB), TB, 0, stream>>>(W_out, Wout_b, D_MODEL * D_INNER);

    // 1) xz = x @ W_in^T  (2048 x 4096 x 1024) — LDS-tiled, async double-buffered
    {
        dim3 grid(N_XZ / BN, MROWS / BM);
        gemm_lds_wmma<<<grid, TB, 0, stream>>>(xb, Win_b, xzbuf, D_MODEL, N_XZ);
    }
    // 2) depthwise conv + SiLU
    conv1d_silu<<<cdiv(MROWS * D_INNER, TB), TB, 0, stream>>>(xzbuf, conv_w, conv_b,
                                                              xconv, xconv_b);
    // 3) x_dbl = x_conv @ W_x^T  (2048 x 96 x 2048) — skinny, direct
    {
        int waves = (MROWS / 16) * (N_DBL / (16 * 2));
        gemm_bf16_wmma<2><<<cdiv(waves, 8), TB, 0, stream>>>(xconv_b, Wx_b, xdbl,
                                                             MROWS, N_DBL, D_INNER, N_DBL);
    }
    // 4) split into dt_low / B / C
    split_xdbl<<<cdiv(MROWS * N_DBL, TB), TB, 0, stream>>>(xdbl, dtlow_b, Bssm, Cssm);
    // 5) dt_raw = dt_low @ W_dt^T  (2048 x 2048 x 64) — K too small for staging
    {
        int waves = (MROWS / 16) * (D_INNER / (16 * 4));
        gemm_bf16_wmma<4><<<cdiv(waves, 8), TB, 0, stream>>>(dtlow_b, Wdt_b, dtbuf,
                                                             MROWS, D_INNER, DT_RANK, D_INNER);
    }
    softplus_dt<<<cdiv(MROWS * D_INNER, TB), TB, 0, stream>>>(dtbuf, b_dt);
    // 6) selective scan + gated epilogue -> y (bf16)
    ssm_scan<<<B_SZ * (D_INNER / 256), 256, 0, stream>>>(dtbuf, Bssm, Cssm, xconv,
                                                         xzbuf, A_log, D_param, y_b);
    // 7) out = y @ W_out^T  (2048 x 1024 x 2048) — LDS-tiled, async double-buffered
    {
        dim3 grid(D_MODEL / BN, MROWS / BM);
        gemm_lds_wmma<<<grid, TB, 0, stream>>>(y_b, Wout_b, out, D_INNER, D_MODEL);
    }
}
